// GenODE_67121748902130
// MI455X (gfx1250) — matrix operand order repaired and checked by reading, not compiled, not run on hardware
//
#include <hip/hip_runtime.h>
#include <hip/hip_bf16.h>

// ---------------------------------------------------------------------------
// Types for CDNA5 WMMA (wave32): v_wmma_f32_16x16x32_bf16
// ---------------------------------------------------------------------------
typedef __bf16 bf16_t;
typedef __attribute__((ext_vector_type(16))) bf16_t v16bf;
typedef __attribute__((ext_vector_type(8)))  bf16_t v8bf;
typedef __attribute__((ext_vector_type(8)))  float  v8f;
typedef __attribute__((ext_vector_type(4)))  int    v4i;

#define BATCH  1024
#define LATENT 64
#define UNITS  512
#define TPTS   16
#define BN     (BATCH * LATENT)          // 65536 floats per state

// CDNA5 async global->LDS copies (ASYNCcnt path), guarded for portability.
#if defined(__has_builtin)
#if __has_builtin(__builtin_amdgcn_global_load_async_to_lds_b128) && \
    __has_builtin(__builtin_amdgcn_s_wait_asynccnt)
#define HAVE_ASYNC_LDS 1
#endif
#endif

// ---------------------------------------------------------------------------
// Dormand-Prince 8(5,3) tableau (exact rationals, evaluated at compile time)
// ---------------------------------------------------------------------------
#define FR(a, b) ((float)((double)(a) / (double)(b)))

__constant__ float c_butcherA[13][13] = {
  {0.f},
  {FR(1,18)},
  {FR(1,48), FR(1,16)},
  {FR(1,32), 0.f, FR(3,32)},
  {FR(5,16), 0.f, FR(-75,64), FR(75,64)},
  {FR(3,80), 0.f, 0.f, FR(3,16), FR(3,20)},
  {FR(29443841,614563906), 0.f, 0.f, FR(77736538,692538347),
   FR(-28693883,1125000000), FR(23124283,1800000000)},
  {FR(16016141,946692911), 0.f, 0.f, FR(61564180,158732637),
   FR(22789713,633445777), FR(545815736,2771057229), FR(-180193667,1043307555)},
  {FR(39632708,573591083), 0.f, 0.f, FR(-433636366,683701615),
   FR(-421739975,2616292301), FR(100302831,723423059),
   FR(790204164,839813087), FR(800635310,3783071287)},
  {FR(246121993,1340847787), 0.f, 0.f, FR(-37695042795.0,15268766246.0),
   FR(-309121744,1061227803), FR(-12992083,490766935),
   FR(6005943493.0,2108947869), FR(393006217,1396673457), FR(123872331,1001029789)},
  {FR(-1028468189,846180014), 0.f, 0.f, FR(8478235783.0,508512852),
   FR(1311729495,1432422823), FR(-10304129995.0,1701304382),
   FR(-48777925059.0,3047939560), FR(15336726248.0,1032824649),
   FR(-45442868181.0,3398467696), FR(3065993473.0,597172653)},
  {FR(185892177,718116043), 0.f, 0.f, FR(-3185094517.0,667107341),
   FR(-477755414,1098053517), FR(-703635378,230739211),
   FR(5731566787.0,1027545527), FR(5232866602.0,850066563),
   FR(-4093664535.0,808688257), FR(3962137247.0,1805957418), FR(65686358,487910083)},
  {FR(403863854,491063109), 0.f, 0.f, FR(-5068492393.0,434740067),
   FR(-411421997,543043805), FR(652783627,914296604),
   FR(11173962825.0,925320556), FR(-13158990841.0,6184727034),
   FR(3936647629.0,1978049680), FR(-160528059,685178525),
   FR(248638103,1413531060), 0.f},
};

__constant__ float c_sol[13] = {
  FR(14005451,335480064), 0.f, 0.f, 0.f, 0.f,
  FR(-59238493,1068277825), FR(181606767,758867731), FR(561292985,797845732),
  FR(-1041891430,1371343529), FR(760417239,1151165299), FR(118820643,751138087),
  FR(-528747749,2220607170), 0.25f,
};

// ---------------------------------------------------------------------------
// Helpers
// ---------------------------------------------------------------------------
static __device__ __forceinline__ unsigned short f2bf(float f) {
  unsigned int u = __float_as_uint(f);
  u += 0x7FFFu + ((u >> 16) & 1u);       // round-to-nearest-even
  return (unsigned short)(u >> 16);
}

static __device__ __forceinline__ float tanh_approx(float x) {
#if defined(__has_builtin)
#if __has_builtin(__builtin_amdgcn_tanhf)
  return __builtin_amdgcn_tanhf(x);      // gfx1250 V_TANH_F32
#else
  return tanhf(x);
#endif
#else
  return tanhf(x);
#endif
}

static __device__ __forceinline__ v16bf cat8(v8bf lo, v8bf hi) {
  v16bf r;
#pragma unroll
  for (int i = 0; i < 8; i++) { r[i] = lo[i]; r[i + 8] = hi[i]; }
  return r;
}

// 16-byte global->LDS copy: async on CDNA5, synchronous fallback otherwise.
static __device__ __forceinline__ void cp16(const unsigned short* g,
                                            unsigned short* l) {
#if defined(HAVE_ASYNC_LDS)
  __builtin_amdgcn_global_load_async_to_lds_b128(
      (__attribute__((address_space(1))) v4i*)g,
      (__attribute__((address_space(3))) v4i*)l, 0, 0);
#else
  *(uint4*)l = *(const uint4*)g;
#endif
}

static __device__ __forceinline__ void wait_async_pf() {
#if defined(HAVE_ASYNC_LDS)
  __builtin_amdgcn_s_wait_asynccnt(6);   // allow next chunk's 6 copies in flight
#endif
}
static __device__ __forceinline__ void wait_async_all() {
#if defined(HAVE_ASYNC_LDS)
  __builtin_amdgcn_s_wait_asynccnt(0);
#endif
}

__global__ void f32_to_bf16_kernel(const float* __restrict__ in,
                                   unsigned short* __restrict__ out, int n) {
  int i = blockIdx.x * blockDim.x + threadIdx.x;
  if (i < n) out[i] = f2bf(in[i]);
}

// ---------------------------------------------------------------------------
// WMMA GEMM:  X = A[M,K](bf16) @ W[K,N](bf16) + bias
//   mode 0: C = X            (fp32 out, no activation)   -> RK k-buffers
//   mode 1: C = bf16(tanh(X))                            -> next-layer input
// Block = 128 threads (4 waves), tile 128x64; wave tile 32x64; K-chunk 32.
// Double-buffered LDS fed by async global->LDS b128 copies.
// Requires M%128==0, N%64==0, K%32==0 (true for all ODE layers).
// ---------------------------------------------------------------------------
__global__ __launch_bounds__(128) void wmma_gemm_kernel(
    const unsigned short* __restrict__ A, const unsigned short* __restrict__ W,
    const float* __restrict__ bias, void* __restrict__ Cout,
    int M, int N, int K, int mode) {
  (void)M;
  // Row strides padded so per-lane b128 fragment loads hit distinct
  // quad-bank groups: 40*2=80B and 72*2=144B are odd multiples of 16B.
  __shared__ unsigned short sA[2][128][40];  // 128 rows x 32 k (bf16 bits)
  __shared__ unsigned short sB[2][32][72];   // 32 k x 64 cols  (bf16 bits)

  const int tid  = threadIdx.x;
  const int wave = tid >> 5;
  const int lane = tid & 31;
  const int half = lane >> 4;
  const int ml   = lane & 15;
  const int rowBase = blockIdx.y * 128;
  const int colBase = blockIdx.x * 64;

  v8f acc[2][4];
#pragma unroll
  for (int g = 0; g < 2; g++)
#pragma unroll
    for (int nt = 0; nt < 4; nt++)
#pragma unroll
      for (int r = 0; r < 8; r++) acc[g][nt][r] = 0.0f;

  // Stage one 32-deep K-chunk into LDS buffer `buf` (6 b128 copies/thread).
  auto stage_copy = [&](int buf, int k0) {
#pragma unroll
    for (int i = 0; i < 4; i++) {            // A: 512 x 16B chunks
      int idx = tid + i * 128;
      int r   = idx >> 2;
      int sub = idx & 3;
      cp16(A + (size_t)(rowBase + r) * K + k0 + sub * 8, &sA[buf][r][sub * 8]);
    }
#pragma unroll
    for (int i = 0; i < 2; i++) {            // W: 256 x 16B chunks
      int idx = tid + i * 128;
      int r   = idx >> 3;
      int sub = idx & 7;
      cp16(W + (size_t)(k0 + r) * N + colBase + sub * 8, &sB[buf][r][sub * 8]);
    }
  };

  const int nk = K >> 5;
  stage_copy(0, 0);
  for (int kc = 0; kc < nk; kc++) {
    const int cur = kc & 1;
    if (kc + 1 < nk) {                       // prefetch next chunk
      stage_copy(cur ^ 1, (kc + 1) << 5);
      wait_async_pf();                       // current chunk's copies landed
    } else {
      wait_async_all();
    }
    __syncthreads();

    // A fragments (ISA 16-bit A 16x32 layout): lane holds row ml,
    // elements 0-7 -> K = 8*half.., elements 8-15 -> +16.
    const unsigned short* pa0 = &sA[cur][wave * 32 + ml][8 * half];
    const unsigned short* pa1 = &sA[cur][wave * 32 + 16 + ml][8 * half];
    v16bf a0 = cat8(*(const v8bf*)pa0, *(const v8bf*)(pa0 + 16));
    v16bf a1 = cat8(*(const v8bf*)pa1, *(const v8bf*)(pa1 + 16));

#pragma unroll
    for (int nt = 0; nt < 4; nt++) {
      // B fragment (32x16): lane holds column N=ml, K = 8*half.. and +16.
      v16bf b;
#pragma unroll
      for (int s = 0; s < 16; s++)
        b[s] = __builtin_bit_cast(bf16_t, sB[cur][(s < 8 ? s : s + 8) + 8 * half][nt * 16 + ml]);
      acc[0][nt] = __builtin_amdgcn_wmma_f32_16x16x32_bf16(
          false, a0, false, b, (short)0, acc[0][nt], false, false);
      acc[1][nt] = __builtin_amdgcn_wmma_f32_16x16x32_bf16(
          false, a1, false, b, (short)0, acc[1][nt], false, false);
    }
    __syncthreads();
  }

  // Epilogue: bias (+ tanh -> bf16), scatter per ISA C/D layout.
#pragma unroll
  for (int nt = 0; nt < 4; nt++) {
    int col = colBase + nt * 16 + ml;
    float bv = bias[col];
#pragma unroll
    for (int g = 0; g < 2; g++) {
#pragma unroll
      for (int r = 0; r < 8; r++) {
        int row = rowBase + wave * 32 + g * 16 + r + 8 * half;
        float v = acc[g][nt][r] + bv;
        if (mode == 1) {
          ((unsigned short*)Cout)[(size_t)row * N + col] = f2bf(tanh_approx(v));
        } else {
          ((float*)Cout)[(size_t)row * N + col] = v;
        }
      }
    }
  }
}

// ---------------------------------------------------------------------------
// Stage input:  yi = bf16( y + dt * sum_{j<i} A[i][j] * k_j )   (fp32 math)
// ---------------------------------------------------------------------------
__global__ void stage_input_kernel(const float* __restrict__ y,
                                   const float* __restrict__ kbuf,
                                   unsigned short* __restrict__ yi,
                                   const float* __restrict__ ts,
                                   int s, int stage) {
  int idx = blockIdx.x * blockDim.x + threadIdx.x;
  if (idx >= BN) return;
  float dt  = ts[s + 1] - ts[s];
  float acc = y[idx];
  for (int j = 0; j < stage; j++) {
    float a = c_butcherA[stage][j];
    if (a != 0.0f) acc += dt * a * kbuf[(size_t)j * BN + idx];
  }
  yi[idx] = f2bf(acc);
}

// y_next = y + dt * sum_i c_sol[i] * k_i      (fp32)
__global__ void rk_combine_kernel(const float* __restrict__ y,
                                  const float* __restrict__ kbuf,
                                  float* __restrict__ ynext,
                                  const float* __restrict__ ts, int s) {
  int idx = blockIdx.x * blockDim.x + threadIdx.x;
  if (idx >= BN) return;
  float dt  = ts[s + 1] - ts[s];
  float acc = y[idx];
#pragma unroll
  for (int i = 0; i < 13; i++) {
    float c = c_sol[i];
    if (c != 0.0f) acc += dt * c * kbuf[(size_t)i * BN + idx];
  }
  ynext[idx] = acc;
}

// ---------------------------------------------------------------------------
// Naive fp32 layer for the tiny aug/dec MLPs (coalesced on W).
// act: 0 none, 1 relu, 2 tanh.  transpose_out: write [T,B] row t*B+b to b*T+t.
// ---------------------------------------------------------------------------
__global__ void mlp_layer_naive_kernel(const float* __restrict__ A,
                                       const float* __restrict__ W,
                                       const float* __restrict__ b,
                                       float* __restrict__ C,
                                       int M, int K, int N, int act,
                                       int transpose_out, int Bdim, int Tdim) {
  int idx = blockIdx.x * blockDim.x + threadIdx.x;
  if (idx >= M * N) return;
  int row = idx / N;
  int col = idx - row * N;
  const float* a = A + (size_t)row * K;
  float s = b[col];
  for (int k = 0; k < K; k++) s += a[k] * W[(size_t)k * N + col];
  if (act == 1)      s = fmaxf(s, 0.0f);
  else if (act == 2) s = tanh_approx(s);
  size_t o;
  if (transpose_out) {
    int t  = row / Bdim;
    int bb = row - t * Bdim;
    o = ((size_t)bb * Tdim + t) * N + col;
  } else {
    o = (size_t)row * N + col;
  }
  C[o] = s;
}

// ---------------------------------------------------------------------------
// Host orchestration
// ---------------------------------------------------------------------------
extern "C" void kernel_launch(void* const* d_in, const int* in_sizes, int n_in,
                              void* d_out, int out_size, void* d_ws, size_t ws_size,
                              hipStream_t stream) {
  (void)in_sizes; (void)n_in; (void)out_size; (void)ws_size;

  // Input leaves in setup_inputs() order (list-of-(W,b) flattens in order).
  const float* x  = (const float*)d_in[0];   // [1024,3]
  const float* ts = (const float*)d_in[1];   // [16]
  const float* augW[4] = {(const float*)d_in[2], (const float*)d_in[4],
                          (const float*)d_in[6], (const float*)d_in[8]};
  const float* augB[4] = {(const float*)d_in[3], (const float*)d_in[5],
                          (const float*)d_in[7], (const float*)d_in[9]};
  const float* odeW[4] = {(const float*)d_in[10], (const float*)d_in[12],
                          (const float*)d_in[14], (const float*)d_in[16]};
  const float* odeB[4] = {(const float*)d_in[11], (const float*)d_in[13],
                          (const float*)d_in[15], (const float*)d_in[17]};
  const float* decW[4] = {(const float*)d_in[18], (const float*)d_in[20],
                          (const float*)d_in[22], (const float*)d_in[24]};
  const float* decB[4] = {(const float*)d_in[19], (const float*)d_in[21],
                          (const float*)d_in[23], (const float*)d_in[25]};
  float* out = (float*)d_out;

  // Carve workspace (256B aligned).
  char* ws = (char*)d_ws;
  size_t off = 0;
  auto carve = [&](size_t bytes) -> void* {
    void* p = ws + off;
    off = (off + bytes + 255) & ~(size_t)255;
    return p;
  };
  const int odeKN[4][2] = {{LATENT, UNITS}, {UNITS, UNITS},
                           {UNITS, UNITS}, {UNITS, LATENT}};
  unsigned short* odeWbf[4];
  for (int l = 0; l < 4; l++)
    odeWbf[l] = (unsigned short*)carve((size_t)odeKN[l][0] * odeKN[l][1] * 2);

  float*          sol  = (float*)carve((size_t)TPTS * BN * 4);   // [T][B][64]
  float*          kbuf = (float*)carve((size_t)13 * BN * 4);     // k_0..k_12
  unsigned short* yi   = (unsigned short*)carve((size_t)BN * 2);            // bf16
  unsigned short* h1   = (unsigned short*)carve((size_t)BATCH * UNITS * 2); // bf16
  unsigned short* h2   = (unsigned short*)carve((size_t)BATCH * UNITS * 2);
  unsigned short* h3   = (unsigned short*)carve((size_t)BATCH * UNITS * 2);
  float* a1 = (float*)carve((size_t)BATCH * 32 * 4);
  float* a2 = (float*)carve((size_t)BATCH * 32 * 4);
  float* a3 = (float*)carve((size_t)BATCH * 32 * 4);
  const int DR = TPTS * BATCH;                                   // 16384 rows
  float* dd1 = (float*)carve((size_t)DR * 100 * 4);
  float* dd2 = (float*)carve((size_t)DR * 100 * 4);
  float* dd3 = (float*)carve((size_t)DR * 100 * 4);

  // 1) Convert ODE weights fp32 -> bf16 (deterministic, re-run every call).
  for (int l = 0; l < 4; l++) {
    int n = odeKN[l][0] * odeKN[l][1];
    f32_to_bf16_kernel<<<(n + 255) / 256, 256, 0, stream>>>(odeW[l], odeWbf[l], n);
  }

  // 2) Aug encoder: 3 -> 32 -> 32 -> 32 -> 64 (ReLU), y0 -> sol[0].
  auto naive = [&](const float* A, const float* W, const float* B, float* C,
                   int M, int K, int N, int act, int tr) {
    int threads = M * N;
    mlp_layer_naive_kernel<<<(threads + 255) / 256, 256, 0, stream>>>(
        A, W, B, C, M, K, N, act, tr, BATCH, TPTS);
  };
  naive(x,  augW[0], augB[0], a1,  BATCH, 3,  32, 1, 0);
  naive(a1, augW[1], augB[1], a2,  BATCH, 32, 32, 1, 0);
  naive(a2, augW[2], augB[2], a3,  BATCH, 32, 32, 1, 0);
  naive(a3, augW[3], augB[3], sol, BATCH, 32, 64, 0, 0);

  // 3) dopri8 integration: 15 steps x 13 stages, f = 64->512->512->512->64.
  auto gemm = [&](const unsigned short* A, const unsigned short* W,
                  const float* B, void* C, int M, int N, int K, int mode) {
    dim3 grid(N / 64, M / 128);
    wmma_gemm_kernel<<<grid, 128, 0, stream>>>(A, W, B, C, M, N, K, mode);
  };
  const int EB = (BN + 255) / 256;
  for (int s = 0; s < TPTS - 1; s++) {
    float* y = sol + (size_t)s * BN;
    for (int stage = 0; stage < 13; stage++) {
      stage_input_kernel<<<EB, 256, 0, stream>>>(y, kbuf, yi, ts, s, stage);
      gemm(yi, odeWbf[0], odeB[0], h1, BATCH, UNITS, LATENT, 1);
      gemm(h1, odeWbf[1], odeB[1], h2, BATCH, UNITS, UNITS,  1);
      gemm(h2, odeWbf[2], odeB[2], h3, BATCH, UNITS, UNITS,  1);
      gemm(h3, odeWbf[3], odeB[3], kbuf + (size_t)stage * BN,
           BATCH, LATENT, UNITS, 0);
    }
    rk_combine_kernel<<<EB, 256, 0, stream>>>(y, kbuf, sol + (size_t)(s + 1) * BN,
                                              ts, s);
  }

  // 4) Decoder on all T*B rows: 64 -> 100 -> 100 -> 100 -> 1 (tanh hidden),
  //    final layer writes transposed [B,T] output.
  naive(sol, decW[0], decB[0], dd1, DR, 64,  100, 2, 0);
  naive(dd1, decW[1], decB[1], dd2, DR, 100, 100, 2, 0);
  naive(dd2, decW[2], decB[2], dd3, DR, 100, 100, 2, 0);
  naive(dd3, decW[3], decB[3], out, DR, 100, 1,   0, 1);
}
